// PEER_59588376264731
// MI455X (gfx1250) — matrix operand order repaired and checked by reading, not compiled, not run on hardware
//
#include <hip/hip_runtime.h>
#include <hip/hip_bf16.h>
#include <math.h>

// ---------------- problem constants (from reference setup) ----------------
#define BATCH   2
#define TLEN    2048
#define DMODEL  1024
#define NHEAD   8
#define NSUB    512          // S
#define HD      128          // head dim
#define KSEL    16           // top_k
#define MROWS   (BATCH*TLEN) // 4096 token rows

typedef __bf16 bf16_t;
typedef __bf16 v16bf __attribute__((ext_vector_type(16)));
typedef __bf16 v8bf  __attribute__((ext_vector_type(8)));
typedef float  v8f   __attribute__((ext_vector_type(8)));
typedef unsigned int v4u __attribute__((ext_vector_type(4)));
typedef int v8i __attribute__((ext_vector_type(8)));
typedef int v4i __attribute__((ext_vector_type(4)));

// ---------------- f32 -> bf16 converters ----------------
__global__ void cvt_f32_bf16(const float* __restrict__ in, bf16_t* __restrict__ out, int n) {
    int i = blockIdx.x * blockDim.x + threadIdx.x;
    if (i < n) out[i] = (bf16_t)in[i];
}

// out[c*R + r] = in[r*C + c]  (weights become k-major: Wt[k][n] = W[n][k])
__global__ void cvt_f32_bf16_T(const float* __restrict__ in, bf16_t* __restrict__ out,
                               int R, int C) {
    int i = blockIdx.x * blockDim.x + threadIdx.x;
    if (i < R * C) {
        const int r = i / C, c = i % C;
        out[(size_t)c * R + r] = (bf16_t)in[i];
    }
}

// ---------------- TDM: 2D tile (rows x elems, 2-byte elements) -> LDS ----------------
// D# layout per CDNA5 ISA ch.8: group0 = {flags, lds_addr, global_addr, type},
// group1 = {mode/data_size, tensor dims, tile dims, dim0 stride}.
__device__ __forceinline__ void tdm_load_tile_2d(unsigned lds_addr,
                                                 const void* gptr,
                                                 unsigned tensor_d0,             // row length remaining (elems)
                                                 unsigned tensor_d1,             // rows remaining
                                                 unsigned long long d0_stride,   // row stride (elems)
                                                 unsigned tile_d0,               // tile width (elems)
                                                 unsigned tile_d1) {             // tile rows
    const unsigned long long ga = (unsigned long long)gptr;
    v4u g0;
    g0.x = 1u;                                                  // count=1 (user descriptor)
    g0.y = lds_addr;                                            // bits 63:32 = LDS byte address
    g0.z = (unsigned)(ga & 0xffffffffu);                        // global_addr[31:0]
    g0.w = (unsigned)((ga >> 32) & 0x01ffffffu) | (2u << 30);   // addr[56:32] | type=2 ("image")
    v8i g1;
    g1[0] = (int)(1u << 16);                                    // workgroup_mask=0, data_size=1 (2B)
    g1[1] = (int)((tensor_d0 & 0xffffu) << 16);                 // tensor_dim0[15:0]  @bits 63:48
    g1[2] = (int)((tensor_d0 >> 16) | ((tensor_d1 & 0xffffu) << 16)); // dim0 hi | dim1 lo
    g1[3] = (int)((tensor_d1 >> 16) | ((tile_d0 & 0xffffu) << 16));   // dim1 hi | tile_dim0
    g1[4] = (int)(tile_d1 & 0xffffu);                           // tile_dim1 (tile_dim2=0 -> unused)
    g1[5] = (int)(unsigned)(d0_stride & 0xffffffffull);         // tensor_dim0_stride[31:0]
    g1[6] = (int)(unsigned)((d0_stride >> 32) & 0xffffull);     // stride[47:32], dim1_stride=0
    g1[7] = 0;
    const v4i z4 = {0, 0, 0, 0};
    const v8i z8 = {0, 0, 0, 0, 0, 0, 0, 0};
    // amdgpu-toolchain (clang-23) 6-arg form: (g0, g1, g2, g3, g4, cpol)
    __builtin_amdgcn_tensor_load_to_lds(g0, g1, z4, z4, z8, 0);
}

__device__ __forceinline__ unsigned lds_byte_addr(const void* p) {
    // generic pointer to __shared__: low 32 bits carry the LDS offset
    return (unsigned)(unsigned long long)p;
}

// ---------------- WMMA GEMM: C[M,N] = A[M,K] * Wt  (Wt is k-major: Wt[k][n]) --------
// Block = 256 threads = 8 waves computing a 32x64 macro-tile (2x4 grid of 16x16 WMMA
// tiles).  Double-buffered TDM pipeline: per 32-deep k-step, wave 0 waits TENSORcnt
// for the current tiles, everyone barriers, wave 0 immediately DMAs the *next* A/B
// tiles into the other LDS buffer while all 8 waves compute v_wmma_f32_16x16x32_bf16
// from the current one.  A is two ds_load_b128, B 16 strided 2-byte reads (ISA 7.12.2 layouts).
#define BM 32
#define BN 64
#define BK 32

__global__ __launch_bounds__(256)
void gemm_bf16_wmma(const bf16_t* __restrict__ A,
                    const bf16_t* __restrict__ Wt,   // [Kd][N] k-major
                    float* __restrict__ Cf,
                    int M, int N, int Kd) {
    __shared__ __align__(16) bf16_t Abuf[2][BM][BK];   // 2 x 2 KB
    __shared__ __align__(16) bf16_t Bbuf[2][BK][BN];   // 2 x 4 KB (k-major rows)
    const int tid  = threadIdx.x;
    const int wave = tid >> 5;
    const int lane = tid & 31;
    const int n0 = blockIdx.x * BN;   // grid.x = N/BN
    const int m0 = blockIdx.y * BM;   // grid.y = M/BM
    const int wm = wave >> 2;         // 0..1
    const int wn = wave & 3;          // 0..3
    const int lm    = lane & 15;
    const int khalf = (lane >= 16) ? 8 : 0;
    const int nit = Kd / BK;

    // prologue: stage tiles for k-step 0 into buffer 0
    if (wave == 0) {
        tdm_load_tile_2d(lds_byte_addr(&Abuf[0][0][0]), A + (size_t)m0 * Kd,
                         (unsigned)Kd, (unsigned)(M - m0), (unsigned long long)Kd, BK, BM);
        tdm_load_tile_2d(lds_byte_addr(&Bbuf[0][0][0]), Wt + n0,
                         (unsigned)(N - n0), (unsigned)Kd, (unsigned long long)N, BN, BK);
    }

    v8f c = {};
    for (int it = 0; it < nit; ++it) {
        const int cur = it & 1;
        if (wave == 0) __builtin_amdgcn_s_wait_tensorcnt(0);   // tiles(it) landed
        __syncthreads();   // tiles(it) visible; everyone done reading buf[cur^1] (it-1)
        if (wave == 0 && (it + 1) < nit) {
            const int k1 = (it + 1) * BK;
            tdm_load_tile_2d(lds_byte_addr(&Abuf[cur ^ 1][0][0]),
                             A + (size_t)m0 * Kd + k1,
                             (unsigned)(Kd - k1), (unsigned)(M - m0),
                             (unsigned long long)Kd, BK, BM);
            tdm_load_tile_2d(lds_byte_addr(&Bbuf[cur ^ 1][0][0]),
                             Wt + (size_t)k1 * N + n0,
                             (unsigned)(N - n0), (unsigned)(Kd - k1),
                             (unsigned long long)N, BN, BK);
        }

        // A fragment: row = wm*16+lm; halves K = khalf..khalf+7 and 16+khalf..23+khalf
        const bf16_t* arow = &Abuf[cur][wm * 16 + lm][0];
        const v8bf a0 = *(const v8bf*)(arow + khalf);
        const v8bf a1 = *(const v8bf*)(arow + 16 + khalf);
        const v16bf a = __builtin_shufflevector(a0, a1,
                0, 1, 2, 3, 4, 5, 6, 7, 8, 9, 10, 11, 12, 13, 14, 15);
        // B fragment: column n = wn*16 + lm; K = khalf..khalf+7 and 16+khalf..23+khalf (same as A)
        const bf16_t* bcol = &Bbuf[cur][0][wn * 16 + lm];
        v16bf b;
#pragma unroll
        for (int s = 0; s < 8; ++s) {
            b[s]     = bcol[(khalf + s) * BN];
            b[s + 8] = bcol[(16 + khalf + s) * BN];
        }

        c = __builtin_amdgcn_wmma_f32_16x16x32_bf16(false, a, false, b,
                                                    (short)0, c, false, false);
    }

    const int mo = (lane >= 16) ? 8 : 0;
#pragma unroll
    for (int j = 0; j < 8; ++j) {
        const size_t o = (size_t)(m0 + wm * 16 + j + mo) * N + (n0 + wn * 16 + lm);
        Cf[o] = c[j];
    }
}

// ---------------- fused PEER: scores -> top-k -> product top-k -> expert combine ----
// one block of 512 threads per (b*t, head)
__global__ __launch_bounds__(NSUB)
void peer_fused(const float* __restrict__ h,       // [MROWS, DMODEL] f32
                const float* __restrict__ keys_a,  // [H, S, HD]
                const float* __restrict__ keys_b,  // [H, S, HD]
                const float* __restrict__ eu,      // [S*S, HD]
                const float* __restrict__ ev,      // [S*S, HD]
                bf16_t* __restrict__ merged_bf) {  // [MROWS, DMODEL]
    const int bid  = blockIdx.x;          // bt*NHEAD + head
    const int bt   = bid >> 3;
    const int head = bid & 7;
    const int tid  = threadIdx.x;

    __shared__ float hvec[HD];
    __shared__ float sa[NSUB], sb[NSUB];
    __shared__ float redv[NSUB];
    __shared__ int   redi[NSUB];
    __shared__ float topav[KSEL], topbv[KSEL];
    __shared__ int   topai[KSEL], topbi[KSEL];
    __shared__ float pv[KSEL * KSEL];
    __shared__ int   pi[KSEL * KSEL];
    __shared__ float selv[KSEL];
    __shared__ int   selpos[KSEL], seli[KSEL];
    __shared__ float wts[KSEL], gcoef[KSEL];
    __shared__ float dred[HD];

    if (tid < HD) hvec[tid] = h[(size_t)bt * DMODEL + head * HD + tid];
    __syncthreads();

    // ---- sub-key scores (each thread: one sub-key row of keys_a and keys_b) ----
    {
        const float* ka = keys_a + ((size_t)head * NSUB + tid) * HD;
        const float* kb = keys_b + ((size_t)head * NSUB + tid) * HD;
        float accA = 0.f, accB = 0.f;
#pragma unroll 4
        for (int d = 0; d < HD; d += 4) {
            const float4 va = *(const float4*)(ka + d);
            const float4 vb = *(const float4*)(kb + d);
            const float h0 = hvec[d], h1 = hvec[d+1], h2 = hvec[d+2], h3 = hvec[d+3];
            accA += va.x*h0 + va.y*h1 + va.z*h2 + va.w*h3;
            accB += vb.x*h0 + vb.y*h1 + vb.z*h2 + vb.w*h3;
        }
        sa[tid] = accA;
        sb[tid] = accB;
    }
    __syncthreads();

    // ---- top-16 extraction (tie -> lowest index, like jax top_k) ----
    auto topk_extract = [&](float* arr, int n, float* tv, int* ti) {
        for (int k = 0; k < KSEL; ++k) {
            if (tid < n) { redv[tid] = arr[tid]; redi[tid] = tid; }
            __syncthreads();
            for (int off = n >> 1; off >= 1; off >>= 1) {
                if (tid < off) {
                    const float v2 = redv[tid + off];
                    const int   i2 = redi[tid + off];
                    if (v2 > redv[tid] || (v2 == redv[tid] && i2 < redi[tid])) {
                        redv[tid] = v2; redi[tid] = i2;
                    }
                }
                __syncthreads();
            }
            if (tid == 0) {
                tv[k] = redv[0]; ti[k] = redi[0];
                arr[redi[0]] = -3.4e38f;
            }
            __syncthreads();
        }
    };

    topk_extract(sa, NSUB, topav, topai);
    topk_extract(sb, NSUB, topbv, topbi);

    // ---- product-key scores (K*K = 256) ----
    if (tid < KSEL * KSEL) {
        pv[tid] = topav[tid >> 4] + topbv[tid & 15];
        pi[tid] = topai[tid >> 4] * NSUB + topbi[tid & 15];
    }
    __syncthreads();
    topk_extract(pv, KSEL * KSEL, selv, selpos);
    if (tid < KSEL) seli[tid] = pi[selpos[tid]];
    __syncthreads();

    // warm the expert-row gather stream (global_prefetch_b8)
    if (tid < KSEL) {
        __builtin_prefetch(eu + (size_t)seli[tid] * HD, 0, 0);
        __builtin_prefetch(ev + (size_t)seli[tid] * HD, 0, 0);
    }

    // ---- softmax over the 16 selected scores ----
    if (tid == 0) {
        const float mx = selv[0];           // descending -> [0] is max
        float s = 0.f;
        for (int k = 0; k < KSEL; ++k) { wts[k] = __expf(selv[k] - mx); s += wts[k]; }
        const float inv = 1.f / s;
        for (int k = 0; k < KSEL; ++k) wts[k] *= inv;
    }
    __syncthreads();

    // ---- per-expert gate: g_k = w_k * sigmoid(h . u_k) ----
    for (int k = 0; k < KSEL; ++k) {
        const size_t row = (size_t)seli[k] * HD;
        if (tid < HD) dred[tid] = hvec[tid] * eu[row + tid];
        __syncthreads();
        for (int off = HD >> 1; off >= 1; off >>= 1) {
            if (tid < off) dred[tid] += dred[tid + off];
            __syncthreads();
        }
        if (tid == 0) gcoef[k] = wts[k] * (1.f / (1.f + __expf(-dred[0])));
        __syncthreads();
    }

    // ---- weighted combine of v-rows (bf16 out feeds the W_out WMMA GEMM) ----
    if (tid < HD) {
        float acc = 0.f;
#pragma unroll
        for (int k = 0; k < KSEL; ++k)
            acc += gcoef[k] * ev[(size_t)seli[k] * HD + tid];
        merged_bf[(size_t)bt * DMODEL + head * HD + tid] = (bf16_t)acc;
    }
}

// ---------------- LayerNorm over last dim (1024), one block per row ----------------
__global__ __launch_bounds__(256)
void layernorm_rows(const float* __restrict__ proj,
                    const float* __restrict__ gamma,
                    const float* __restrict__ beta,
                    float* __restrict__ out) {
    const int row = blockIdx.x;
    const int tid = threadIdx.x;
    __shared__ float red[256];
    const float* p = proj + (size_t)row * DMODEL;

    float s = 0.f;
    for (int i = tid; i < DMODEL; i += 256) s += p[i];
    red[tid] = s; __syncthreads();
    for (int off = 128; off >= 1; off >>= 1) {
        if (tid < off) red[tid] += red[tid + off];
        __syncthreads();
    }
    const float mu = red[0] * (1.f / DMODEL);
    __syncthreads();

    float v = 0.f;
    for (int i = tid; i < DMODEL; i += 256) { const float d = p[i] - mu; v += d * d; }
    red[tid] = v; __syncthreads();
    for (int off = 128; off >= 1; off >>= 1) {
        if (tid < off) red[tid] += red[tid + off];
        __syncthreads();
    }
    const float inv = rsqrtf(red[0] * (1.f / DMODEL) + 1e-5f);

    for (int i = tid; i < DMODEL; i += 256)
        out[(size_t)row * DMODEL + i] = (p[i] - mu) * inv * gamma[i] + beta[i];
}

// =======================================================================
extern "C" void kernel_launch(void* const* d_in, const int* in_sizes, int n_in,
                              void* d_out, int out_size, void* d_ws, size_t ws_size,
                              hipStream_t stream) {
    (void)in_sizes; (void)n_in; (void)out_size; (void)ws_size;

    const float* x      = (const float*)d_in[0];
    const float* W_in   = (const float*)d_in[1];
    const float* W_out  = (const float*)d_in[2];
    const float* keys_a = (const float*)d_in[3];
    const float* keys_b = (const float*)d_in[4];
    const float* eu     = (const float*)d_in[5];
    const float* ev     = (const float*)d_in[6];
    const float* gamma  = (const float*)d_in[7];
    const float* beta   = (const float*)d_in[8];
    // d_in[9] = top_k (16, compile-time like all shapes)
    float* out = (float*)d_out;

    // ---- workspace carve-up (byte offsets, all 256B aligned) ----
    char* ws = (char*)d_ws;
    size_t off = 0;
    bf16_t* x_bf      = (bf16_t*)(ws + off); off += (size_t)MROWS * DMODEL * 2;  // 8 MB
    bf16_t* WiT_bf    = (bf16_t*)(ws + off); off += (size_t)DMODEL * DMODEL * 2; // 2 MB (k-major)
    bf16_t* WoT_bf    = (bf16_t*)(ws + off); off += (size_t)DMODEL * DMODEL * 2; // 2 MB (k-major)
    float*  h_f       = (float* )(ws + off); off += (size_t)MROWS * DMODEL * 4;  // 16 MB
    bf16_t* merged_bf = (bf16_t*)(ws + off); off += (size_t)MROWS * DMODEL * 2;  // 8 MB
    float*  proj_f    = (float* )(ws + off); off += (size_t)MROWS * DMODEL * 4;  // 16 MB

    // ---- 1) convert x (row-major) and W_in/W_out (transposed to k-major) ----
    {
        const int nx = MROWS * DMODEL;
        const int nw = DMODEL * DMODEL;
        cvt_f32_bf16  <<<(nx + 255) / 256, 256, 0, stream>>>(x, x_bf, nx);
        cvt_f32_bf16_T<<<(nw + 255) / 256, 256, 0, stream>>>(W_in, WiT_bf, DMODEL, DMODEL);
        cvt_f32_bf16_T<<<(nw + 255) / 256, 256, 0, stream>>>(W_out, WoT_bf, DMODEL, DMODEL);
    }

    // ---- 2) h = x @ W_in^T  (double-buffered TDM + WMMA bf16, f32 accum) ----
    {
        dim3 grid(DMODEL / BN, MROWS / BM);   // (16, 128)
        gemm_bf16_wmma<<<grid, 256, 0, stream>>>(x_bf, WiT_bf, h_f,
                                                 MROWS, DMODEL, DMODEL);
    }

    // ---- 3) fused PEER: scores + double top-k + expert combine ----
    peer_fused<<<MROWS * NHEAD, NSUB, 0, stream>>>(h_f, keys_a, keys_b, eu, ev,
                                                   merged_bf);

    // ---- 4) proj = merged @ W_out^T ----
    {
        dim3 grid(DMODEL / BN, MROWS / BM);
        gemm_bf16_wmma<<<grid, 256, 0, stream>>>(merged_bf, WoT_bf, proj_f,
                                                 MROWS, DMODEL, DMODEL);
    }

    // ---- 5) LayerNorm rows -> d_out ----
    layernorm_rows<<<MROWS, 256, 0, stream>>>(proj_f, gamma, beta, out);
}
